// MultiCondLayer_77962246357641
// MI455X (gfx1250) — matrix operand order repaired and checked by reading, not compiled, not run on hardware
//
#include <hip/hip_runtime.h>

// ---------------------------------------------------------------------------
// out[b,o,n] = (sum_c cond[b,c,n] * Wsum[o,c] + bsum[o]) * x_mask[b,0,n]
//   Wsum[o,c] = sum_k W[k,o,c],  bsum[o] = sum_k b[k,o]
// Per batch: GEMM M=1024 (O), N=4096, K=1024 (C).
//
// Split-bf16 ("bf16x3"): x = hi + lo (both RNE bf16);
//   A*B ~= Ahi*Bhi + Ahi*Blo + Alo*Bhi, f32 accumulate via
//   V_WMMA_F32_16X16X32_BF16.  ~2^-17 per-product error -> ~1e-5 overall.
// A tiles (pre-decomposed bf16 in workspace) are moved by the CDNA5 async
// engine (GLOBAL_LOAD_ASYNC_TO_LDS_B128 + s_wait_asynccnt): no VGPR staging.
// B (cond, f32) is staged through 8 VGPRs, split to bf16 hi/lo, and stored
// TRANSPOSED in LDS so fragments are contiguous b128 loads.
// ---------------------------------------------------------------------------

typedef __attribute__((ext_vector_type(16))) __bf16 v16bf;
typedef __attribute__((ext_vector_type(8)))  __bf16 v8bf;
typedef __attribute__((ext_vector_type(4)))  __bf16 v4bf;
typedef __attribute__((ext_vector_type(8)))  float  v8f;

// Builtin-facing types (match the diagnostic: vector_size(16) int, AS1/AS3).
typedef int v4i __attribute__((vector_size(4 * sizeof(int))));
typedef __attribute__((address_space(1))) v4i* g_v4i_ptr;
typedef __attribute__((address_space(3))) v4i* l_v4i_ptr;

#define B_SZ  8
#define C_K   1024
#define N_N   4096
#define O_M   1024

#define BM    128
#define BN    64
#define BK    32
#define LDA   40     // LDS row stride (bf16) = 80B: 16B-aligned, conflict-breaking
#define LDB   40
#define KSLABS (C_K / BK)   // 32

#if __has_builtin(__builtin_amdgcn_global_load_async_to_lds_b128)
#define ASYNC_LDS 1
#endif

// --- helpers ---------------------------------------------------------------
__device__ __forceinline__ void async_b128(const __bf16* g, __bf16* l) {
#ifdef ASYNC_LDS
  // AS-qualified pointers via integer reinterpret: flat LDS addr low 32 bits
  // are the LDS offset (ISA 10.2 aperture rules); global flat == AS1 bits.
  __builtin_amdgcn_global_load_async_to_lds_b128(
      (g_v4i_ptr)(unsigned long long)g,
      (l_v4i_ptr)(unsigned int)(unsigned long long)l,
      0, 0);
#else
  *(uint4*)l = *(const uint4*)g;   // sync fallback
#endif
}

__device__ __forceinline__ void wait_async0() {
#ifdef ASYNC_LDS
#if __has_builtin(__builtin_amdgcn_s_wait_asynccnt)
  __builtin_amdgcn_s_wait_asynccnt(0);
#else
  asm volatile("s_wait_asynccnt 0x0" ::: "memory");
#endif
#endif
}

__device__ __forceinline__ v16bf ld_frag(const __bf16* p0, const __bf16* p1) {
  union { v16bf v; v8bf h[2]; } u;
  u.h[0] = *(const v8bf*)p0;
  u.h[1] = *(const v8bf*)p1;
  return u.v;
}

__device__ __forceinline__ void split_bf16(float x, __bf16& hi, __bf16& lo) {
  hi = (__bf16)x;
  lo = (__bf16)(x - (float)hi);
}

// --- kernel 1: fold W over k, decompose to bf16 hi/lo; fold bias -----------
__global__ __launch_bounds__(256)
void reduce_w_kernel(const float* __restrict__ W, const float* __restrict__ bias,
                     __bf16* __restrict__ Ahi, __bf16* __restrict__ Alo,
                     float* __restrict__ bsum) {
  const int OC4 = (O_M * C_K) / 4;
  const int i = blockIdx.x * blockDim.x + threadIdx.x;
  if (i < OC4) {
    const float4* w4 = (const float4*)W;
    float4 a = w4[i];
    float4 b1 = w4[i + OC4];
    float4 c = w4[i + 2 * OC4];
    float4 d = w4[i + 3 * OC4];
    float s[4];
    s[0] = a.x + b1.x + c.x + d.x;
    s[1] = a.y + b1.y + c.y + d.y;
    s[2] = a.z + b1.z + c.z + d.z;
    s[3] = a.w + b1.w + c.w + d.w;
    v4bf hv, lv;
#pragma unroll
    for (int e = 0; e < 4; ++e) {
      __bf16 h, l;
      split_bf16(s[e], h, l);
      hv[e] = h; lv[e] = l;
    }
    *(v4bf*)(Ahi + 4 * (size_t)i) = hv;
    *(v4bf*)(Alo + 4 * (size_t)i) = lv;
  }
  if (i < O_M) {
    bsum[i] = bias[i] + bias[O_M + i] + bias[2 * O_M + i] + bias[3 * O_M + i];
  }
}

// --- kernel 2: split-bf16 WMMA GEMM with async A staging --------------------
__global__ __launch_bounds__(256)
void MultiCondLayer_77962246357641_kernel(const float* __restrict__ cond,
                                          const float* __restrict__ x_mask,
                                          const __bf16* __restrict__ Ahi_g,
                                          const __bf16* __restrict__ Alo_g,
                                          const float* __restrict__ bsum,
                                          float* __restrict__ out) {
  __shared__ __bf16 Ah[2][BM][LDA];
  __shared__ __bf16 Al[2][BM][LDA];
  __shared__ __bf16 Bh[2][BN][LDB];   // transposed: Bt[n][k]
  __shared__ __bf16 Bl[2][BN][LDB];

  const int bz     = blockIdx.z;
  const int m_base = blockIdx.y * BM;
  const int n_base = blockIdx.x * BN;

  const int tid  = threadIdx.x;
  const int wid  = tid >> 5;
  const int lane = tid & 31;
  const int lr   = lane & 15;
  const int lh   = lane >> 4;

  const int wm = (wid & 3) * 32;
  const int wn = (wid >> 2) * 32;

  v8f acc[2][2] = {};

  // A staging: 2 threads/row, each owns 16 bf16 (32B) of hi and of lo.
  const int a_r = tid >> 1;
  const int a_h = (tid & 1) * 16;
  const __bf16* aHiSrc = Ahi_g + (size_t)(m_base + a_r) * C_K + a_h;
  const __bf16* aLoSrc = Alo_g + (size_t)(m_base + a_r) * C_K + a_h;
  // B staging: thread owns column n = tid&63, k-rows b_k..b_k+7.
  const int b_n = tid & 63;
  const int b_k = (tid >> 6) * 8;
  const float* bSrc = cond + ((size_t)bz * C_K + b_k) * N_N + n_base + b_n;

  float rb[8];

  auto issueA = [&](int s, int bu) {
    const int k0 = s * BK;
    async_b128(aHiSrc + k0,     &Ah[bu][a_r][a_h]);
    async_b128(aHiSrc + k0 + 8, &Ah[bu][a_r][a_h + 8]);
    async_b128(aLoSrc + k0,     &Al[bu][a_r][a_h]);
    async_b128(aLoSrc + k0 + 8, &Al[bu][a_r][a_h + 8]);
  };
  auto fetchB = [&](int s) {
    const float* bp = bSrc + (size_t)s * BK * N_N;
#pragma unroll
    for (int r = 0; r < 8; ++r) rb[r] = bp[(size_t)r * N_N];
  };

  // prologue: slab 0 in flight
  issueA(0, 0);
  fetchB(0);

  int buf = 0;
  for (int s = 0; s < KSLABS; ++s) {
    // split B slab s -> LDS[buf] (transposed)
    {
      v8bf hv, lv;
#pragma unroll
      for (int r = 0; r < 8; ++r) {
        __bf16 h, l;
        split_bf16(rb[r], h, l);
        hv[r] = h; lv[r] = l;
      }
      *(v8bf*)&Bh[buf][b_n][b_k] = hv;
      *(v8bf*)&Bl[buf][b_n][b_k] = lv;
    }
    wait_async0();        // A slab s resident in LDS[buf]
    __syncthreads();      // also orders all compute(s-1) before slab s+1 writes

    if (s + 1 < KSLABS) {
      issueA(s + 1, buf ^ 1);   // async: overlaps with compute below
      fetchB(s + 1);
      if (s + 2 < KSLABS)
        __builtin_prefetch(bSrc + (size_t)(s + 2) * BK * N_N, 0, 1);
    }

    // compute slab s: one K=32 step, 2x2 tiles, 3 WMMAs each.
    // Only one A-fragment pair live at a time to cap VGPR pressure.
#pragma unroll
    for (int i = 0; i < 2; ++i) {
      const __bf16* rowH = &Ah[buf][wm + i * 16 + lr][0];
      const __bf16* rowL = &Al[buf][wm + i * 16 + lr][0];
      v16bf afh = ld_frag(rowH + lh * 8, rowH + 16 + lh * 8);
      v16bf afl = ld_frag(rowL + lh * 8, rowL + 16 + lh * 8);
#pragma unroll
      for (int j = 0; j < 2; ++j) {
        const __bf16* brH = &Bh[buf][wn + j * 16 + lr][0];
        const __bf16* brL = &Bl[buf][wn + j * 16 + lr][0];
        v16bf bfh = ld_frag(brH + 8 * lh, brH + 16 + 8 * lh);
        v16bf bfl = ld_frag(brL + 8 * lh, brL + 16 + 8 * lh);
        acc[i][j] = __builtin_amdgcn_wmma_f32_16x16x32_bf16(
            false, afh, false, bfh, (short)0, acc[i][j], false, false);
        acc[i][j] = __builtin_amdgcn_wmma_f32_16x16x32_bf16(
            false, afh, false, bfl, (short)0, acc[i][j], false, false);
        acc[i][j] = __builtin_amdgcn_wmma_f32_16x16x32_bf16(
            false, afl, false, bfh, (short)0, acc[i][j], false, false);
      }
    }
    buf ^= 1;
  }

  // epilogue: bias + mask; C/D layout: VGPR v -> M = v + 8*lh, N = lr
#pragma unroll
  for (int j = 0; j < 2; ++j) {
    const int n = n_base + wn + j * 16 + lr;
    const float mask = x_mask[(size_t)bz * N_N + n];
#pragma unroll
    for (int i = 0; i < 2; ++i) {
      const int mrow = m_base + wm + i * 16 + lh * 8;
#pragma unroll
      for (int v = 0; v < 8; ++v) {
        const int m = mrow + v;
        out[((size_t)bz * O_M + m) * N_N + n] = (acc[i][j][v] + bsum[m]) * mask;
      }
    }
  }
}

extern "C" void kernel_launch(void* const* d_in, const int* in_sizes, int n_in,
                              void* d_out, int out_size, void* d_ws, size_t ws_size,
                              hipStream_t stream) {
  (void)in_sizes; (void)n_in; (void)out_size; (void)ws_size;

  const float* cond   = (const float*)d_in[0];   // [8,1024,4096]
  const float* x_mask = (const float*)d_in[1];   // [8,1,4096]
  const float* W      = (const float*)d_in[2];   // [4,1024,1024]
  const float* bias   = (const float*)d_in[3];   // [4,1024]
  float* out = (float*)d_out;                    // [8,1024,4096]

  __bf16* Ahi = (__bf16*)d_ws;                           // 2 MB
  __bf16* Alo = Ahi + (size_t)O_M * C_K;                 // 2 MB
  float*  bsum = (float*)(Alo + (size_t)O_M * C_K);      // 4 KB

  {
    const int total = (O_M * C_K) / 4;
    reduce_w_kernel<<<dim3((total + 255) / 256), dim3(256), 0, stream>>>(
        W, bias, Ahi, Alo, bsum);
  }
  {
    dim3 grid(N_N / BN, O_M / BM, B_SZ);   // (64, 8, 8)
    MultiCondLayer_77962246357641_kernel<<<grid, dim3(256), 0, stream>>>(
        cond, x_mask, Ahi, Alo, bsum, out);
  }
}
